// LoRA_55319178772936
// MI455X (gfx1250) — matrix operand order repaired and checked by reading, not compiled, not run on hardware
//
#include <hip/hip_runtime.h>
#include <hip/hip_bf16.h>

// ---------------------------------------------------------------------------
// LoRA fused linear for MI455X (gfx1250, wave32, WMMA)
//   out = x @ (W + (ALPHA/RANK) * B@A)^T + bias
//   GEMM: M=16384, N=2048, K=2048, bf16 inputs, fp32 accumulate.
// v2: LDS-staged GEMM. Each 128x128 block stages 128x64 X and W slabs in LDS
// (loaded once per block -> 3x less L2 traffic than direct-from-global),
// WMMA fragments are fed by ds_load_b128 with a 16B row pad (72-elem stride)
// for bank-conflict-free access. Global loads for slab s+1 are issued before
// the compute phase of slab s, so HBM/L2 latency hides under 16 WMMAs.
// ---------------------------------------------------------------------------

#define D_DIM 2048
#define M_DIM 16384
#define LORA_SCALE 2.0f   // ALPHA/RANK = 32/16
#define BK 64
#define LDS_STRIDE 72     // 64 + 8 bf16 pad (16B) -> conflict-free ds reads
#define NSLAB (D_DIM / BK)

typedef __bf16  bf16x16 __attribute__((ext_vector_type(16)));
typedef __bf16  bf16x8  __attribute__((ext_vector_type(8)));
typedef float   f32x8   __attribute__((ext_vector_type(8)));
typedef float   f32x4   __attribute__((ext_vector_type(4)));
typedef int     i32x4   __attribute__((ext_vector_type(4)));

// ---------------- Kernel 1: x (f32) -> bf16, 8 elems/thread -----------------
__global__ __launch_bounds__(256)
void lora_cvt_x_bf16(const float* __restrict__ x, __bf16* __restrict__ xb, int n) {
    int i = (blockIdx.x * 256 + threadIdx.x) * 8;
    if (i + 8 > n) return;
    f32x4 a = *(const f32x4*)(x + i);
    f32x4 b = *(const f32x4*)(x + i + 4);
    bf16x8 o;
#pragma unroll
    for (int j = 0; j < 4; ++j) { o[j] = (__bf16)a[j]; o[4 + j] = (__bf16)b[j]; }
    *(bf16x8*)(xb + i) = o;
}

// ------- Kernel 2: wb[o,d] = bf16( W[o,d] + 2 * sum_r B[o,r]*A[r,d] ) -------
__global__ __launch_bounds__(256)
void lora_fuse_w_bf16(const float* __restrict__ W, const float* __restrict__ Bm,
                      const float* __restrict__ Am, __bf16* __restrict__ wb) {
    int idx = blockIdx.x * 256 + threadIdx.x;   // D*D/8 threads
    int o = idx >> 8;
    int d = (idx & 255) * 8;

    float br[16];
#pragma unroll
    for (int r = 0; r < 16; ++r) br[r] = Bm[o * 16 + r] * LORA_SCALE;

    f32x4 w0 = *(const f32x4*)(W + (size_t)o * D_DIM + d);
    f32x4 w1 = *(const f32x4*)(W + (size_t)o * D_DIM + d + 4);
    float acc[8];
#pragma unroll
    for (int j = 0; j < 4; ++j) { acc[j] = w0[j]; acc[4 + j] = w1[j]; }

#pragma unroll
    for (int r = 0; r < 16; ++r) {
        f32x4 a0 = *(const f32x4*)(Am + (size_t)r * D_DIM + d);
        f32x4 a1 = *(const f32x4*)(Am + (size_t)r * D_DIM + d + 4);
#pragma unroll
        for (int j = 0; j < 4; ++j) {
            acc[j]     = fmaf(br[r], a0[j], acc[j]);
            acc[4 + j] = fmaf(br[r], a1[j], acc[4 + j]);
        }
    }
    bf16x8 ob;
#pragma unroll
    for (int j = 0; j < 8; ++j) ob[j] = (__bf16)acc[j];
    *(bf16x8*)(wb + (size_t)o * D_DIM + d) = ob;
}

// ------------- Kernel 3: C = Xb * Wb^T + bias, LDS-staged WMMA --------------
// Block: 256 threads (8 waves), tile 128(M) x 128(N), K-slab 64.
// Wave grid 4x2: each wave owns 32x64 = 2x4 tiles of 16x16 -> 16 WMMAs/slab.
__global__ __launch_bounds__(256)
void lora_gemm_wmma(const __bf16* __restrict__ Xb, const __bf16* __restrict__ Wb,
                    const float* __restrict__ bias, float* __restrict__ out) {
    __shared__ __bf16 sX[128 * LDS_STRIDE];   // 18 KB
    __shared__ __bf16 sW[128 * LDS_STRIDE];   // 18 KB

    const int tid  = threadIdx.x;
    const int lane = tid & 31;
    const int wave = tid >> 5;
    const int half = lane >> 4;
    const int l16  = lane & 15;
    const int wm   = wave & 3;      // 4 waves along M
    const int wn   = wave >> 2;     // 2 waves along N
    const int mblk = blockIdx.y * 128;
    const int nblk = blockIdx.x * 128;
    const int m0   = mblk + wm * 32;
    const int n0   = nblk + wn * 64;

    // --- cooperative-load geometry: 1024 chunks of 16B; 4 chunks/thread ----
    int    ldsOff[4];
    size_t gOffX[4], gOffW[4];
#pragma unroll
    for (int i = 0; i < 4; ++i) {
        int c   = tid + i * 256;
        int row = c >> 3;           // 8 chunks of 16B per 64-elem row
        int cir = c & 7;
        ldsOff[i] = row * LDS_STRIDE + cir * 8;
        gOffX[i]  = (size_t)(mblk + row) * D_DIM + cir * 8;
        gOffW[i]  = (size_t)(nblk + row) * D_DIM + cir * 8;
    }

    f32x8 acc[2][4];
#pragma unroll
    for (int i = 0; i < 2; ++i)
#pragma unroll
        for (int j = 0; j < 4; ++j) acc[i][j] = (f32x8)0.0f;

    // Per-lane LDS fragment base pointers (wave32 WMMA layouts, §7.12.2).
    const __bf16* sxa0 = &sX[(wm * 32 + l16) * LDS_STRIDE + half * 8];
    const __bf16* sxa1 = sxa0 + 16 * LDS_STRIDE;
    const __bf16* swb0 = &sW[(wn * 64 + l16) * LDS_STRIDE + half * 8];

    i32x4 rx[4], rw[4];
    // Prologue: slab 0 global -> regs -> LDS.
#pragma unroll
    for (int i = 0; i < 4; ++i) {
        rx[i] = *(const i32x4*)(Xb + gOffX[i]);
        rw[i] = *(const i32x4*)(Wb + gOffW[i]);
    }
#pragma unroll
    for (int i = 0; i < 4; ++i) {
        *(i32x4*)(&sX[ldsOff[i]]) = rx[i];
        *(i32x4*)(&sW[ldsOff[i]]) = rw[i];
    }
    __syncthreads();

    for (int s = 0; s < NSLAB; ++s) {
        const bool have_next = (s + 1) < NSLAB;
        if (have_next) {            // uniform: issue next slab's global loads
            const int kn = (s + 1) * BK;
#pragma unroll
            for (int i = 0; i < 4; ++i) {
                rx[i] = *(const i32x4*)(Xb + gOffX[i] + kn);
                rw[i] = *(const i32x4*)(Wb + gOffW[i] + kn);
            }
            __builtin_prefetch(Xb + gOffX[0] + kn + BK, 0, 1);
            __builtin_prefetch(Wb + gOffW[0] + kn + BK, 0, 1);
        }

        // ---- compute: two K-steps of 32 from LDS, 8 WMMAs each ----
#pragma unroll
        for (int kk = 0; kk < BK; kk += 32) {
            bf16x16 afrag[2], bfrag[4];
            {
                bf16x8 c0 = *(const bf16x8*)(sxa0 + kk);
                bf16x8 c1 = *(const bf16x8*)(sxa0 + kk + 16);
                afrag[0] = __builtin_shufflevector(c0, c1, 0,1,2,3,4,5,6,7,8,9,10,11,12,13,14,15);
                bf16x8 c2 = *(const bf16x8*)(sxa1 + kk);
                bf16x8 c3 = *(const bf16x8*)(sxa1 + kk + 16);
                afrag[1] = __builtin_shufflevector(c2, c3, 0,1,2,3,4,5,6,7,8,9,10,11,12,13,14,15);
            }
#pragma unroll
            for (int t = 0; t < 4; ++t) {
                bf16x8 b0 = *(const bf16x8*)(swb0 + t * 16 * LDS_STRIDE + kk);
                bf16x8 b1 = *(const bf16x8*)(swb0 + t * 16 * LDS_STRIDE + kk + 16);
                bfrag[t] = __builtin_shufflevector(b0, b1, 0,1,2,3,4,5,6,7,8,9,10,11,12,13,14,15);
            }
#pragma unroll
            for (int i = 0; i < 2; ++i)
#pragma unroll
                for (int j = 0; j < 4; ++j)
                    acc[i][j] = __builtin_amdgcn_wmma_f32_16x16x32_bf16(
                        false, afrag[i], false, bfrag[j],
                        (short)0, acc[i][j], false, false);
        }

        __syncthreads();            // all waves done reading this slab
        if (have_next) {            // uniform
#pragma unroll
            for (int i = 0; i < 4; ++i) {
                *(i32x4*)(&sX[ldsOff[i]]) = rx[i];
                *(i32x4*)(&sW[ldsOff[i]]) = rw[i];
            }
            __syncthreads();        // slab s+1 visible to all waves
        }
    }

    // Epilogue: C tile layout — VGPR v, lane<16 -> (M=v, N=lane); lane>=16 ->
    // (M=8+v, N=lane-16).  N stride of out == D_DIM.
#pragma unroll
    for (int j = 0; j < 4; ++j) {
        const float bv = bias[n0 + j * 16 + l16];
#pragma unroll
        for (int i = 0; i < 2; ++i) {
            float* po = out + (size_t)(m0 + i * 16 + half * 8) * D_DIM + n0 + j * 16 + l16;
#pragma unroll
            for (int v = 0; v < 8; ++v)
                po[(size_t)v * D_DIM] = acc[i][j][v] + bv;
        }
    }
}

// ---------------------------------------------------------------------------
extern "C" void kernel_launch(void* const* d_in, const int* in_sizes, int n_in,
                              void* d_out, int out_size, void* d_ws, size_t ws_size,
                              hipStream_t stream) {
    const float* x  = (const float*)d_in[0];   // [8,2048,2048]
    const float* W  = (const float*)d_in[1];   // [2048,2048]
    const float* b  = (const float*)d_in[2];   // [2048]
    const float* Bm = (const float*)d_in[3];   // [2048,16]
    const float* Am = (const float*)d_in[4];   // [16,2048]
    float* out = (float*)d_out;

    const int nx = in_sizes[0];                        // 33,554,432
    __bf16* xb = (__bf16*)d_ws;                        // 64 MB
    __bf16* wb = (__bf16*)((char*)d_ws + (size_t)nx * sizeof(__bf16)); // +8 MB

    lora_cvt_x_bf16<<<nx / (256 * 8), 256, 0, stream>>>(x, xb, nx);
    lora_fuse_w_bf16<<<(D_DIM * D_DIM / 8) / 256, 256, 0, stream>>>(W, Bm, Am, wb);

    dim3 grid(D_DIM / 128, M_DIM / 128);               // (16, 128)
    lora_gemm_wmma<<<grid, 256, 0, stream>>>(xb, wb, b, out);
}
